// AtomGraphEncoder_21620865368732
// MI455X (gfx1250) — matrix operand order repaired and checked by reading, not compiled, not run on hardware
//
#include <hip/hip_runtime.h>

#define NN 100000
#define EE 400000
#define AD 74
#define KIN 96          // AD padded to multiple of 32
#define HID 256
#define NHEAD 4
#define DHEAD 64

typedef __attribute__((ext_vector_type(16))) __bf16 v16bf;
typedef __attribute__((ext_vector_type(8)))  float  v8f;

union Frag16 { v16bf v; uint4 u[2]; };
union F8 { float4 v[2]; float f[8]; };

// ---------------------------------------------------------------------------
// Utility kernels
// ---------------------------------------------------------------------------

// atom_feat [N,74] fp32 -> [N,96] bf16, zero padded
__global__ void pad_atom_kernel(const float* __restrict__ atom, __bf16* __restrict__ abf) {
    int i = blockIdx.x * blockDim.x + threadIdx.x;          // N*96 threads
    int row = i / KIN, c = i - row * KIN;
    abf[i] = (c < AD) ? (__bf16)atom[(size_t)row * AD + c] : (__bf16)0.0f;
}

// W [K,256] fp32 -> Wt [256,Kpad] bf16 (transposed, K-major, zero padded)
__global__ void prep_wt_kernel(const float* __restrict__ W, __bf16* __restrict__ Wt,
                               int K, int Kpad) {
    int i = blockIdx.x * blockDim.x + threadIdx.x;          // 256*Kpad threads
    int col = i / Kpad, k = i - col * Kpad;
    Wt[i] = (k < K) ? (__bf16)W[(size_t)k * HID + col] : (__bf16)0.0f;
}

__global__ void cast_bf16_kernel(const float* __restrict__ in, __bf16* __restrict__ o) {
    int i = blockIdx.x * blockDim.x + threadIdx.x;          // N*256 threads
    o[i] = (__bf16)in[i];
}

// init segment-max to -inf, denom to 0, output to broadcast bias
__global__ void init_layer_kernel(float* __restrict__ mbuf, float* __restrict__ denom,
                                  float* __restrict__ outb, const float* __restrict__ bout) {
    int i = blockIdx.x * blockDim.x + threadIdx.x;          // N*256 threads
    outb[i] = bout[i & (HID - 1)];
    if (i < NN * NHEAD) { mbuf[i] = -__builtin_inff(); denom[i] = 0.0f; }
}

// ---------------------------------------------------------------------------
// WMMA GEMM: C[M,256] = A[M,K](bf16) * Wt[256,K](bf16, K-major) + bias
// grid.x = M/32 ; block = 256 threads = 8 waves
// wave w: rows [32*blockIdx.x, +32), cols [32w, 32w+32)  (2x2 tiles of 16x16)
// ---------------------------------------------------------------------------
__global__ void gemm_bias_wmma(const __bf16* __restrict__ A, const __bf16* __restrict__ Wt,
                               const float* __restrict__ bias, float* __restrict__ C,
                               int K) {
    const int wave = threadIdx.x >> 5;
    const int lane = threadIdx.x & 31;
    const int hf   = lane >> 4;       // lane half
    const int lm   = lane & 15;
    const int mbase = blockIdx.x * 32;
    const int n0 = wave * 32;
    const int n1 = n0 + 16;

    // A fragment: lanes 0-15 hold K {k..k+7, k+16..k+23},
    //             lanes 16-31 hold K {k+8..k+15, k+24..k+31}
    const __bf16* a0 = A  + (size_t)(mbase + lm) * K + hf * 8;
    const __bf16* a1 = a0 + (size_t)16 * K;
    // B fragment: lanes 0-15 hold K {k..k+7, k+16..k+23},
    //             lanes 16-31 hold K {k+8..k+15, k+24..k+31}
    const __bf16* b0 = Wt + (size_t)(n0 + lm) * K + hf * 8;
    const __bf16* b1 = b0 + (size_t)16 * K;

    v8f acc00 = {}, acc01 = {}, acc10 = {}, acc11 = {};
    for (int k = 0; k < K; k += 32) {
        Frag16 af0, af1, bf0, bf1;
        af0.u[0] = *(const uint4*)(a0 + k);
        af0.u[1] = *(const uint4*)(a0 + k + 16);
        af1.u[0] = *(const uint4*)(a1 + k);
        af1.u[1] = *(const uint4*)(a1 + k + 16);
        bf0.u[0] = *(const uint4*)(b0 + k);
        bf0.u[1] = *(const uint4*)(b0 + k + 16);
        bf1.u[0] = *(const uint4*)(b1 + k);
        bf1.u[1] = *(const uint4*)(b1 + k + 16);
        acc00 = __builtin_amdgcn_wmma_f32_16x16x32_bf16(false, af0.v, false, bf0.v,
                                                        (short)0, acc00, false, false);
        acc01 = __builtin_amdgcn_wmma_f32_16x16x32_bf16(false, af0.v, false, bf1.v,
                                                        (short)0, acc01, false, false);
        acc10 = __builtin_amdgcn_wmma_f32_16x16x32_bf16(false, af1.v, false, bf0.v,
                                                        (short)0, acc10, false, false);
        acc11 = __builtin_amdgcn_wmma_f32_16x16x32_bf16(false, af1.v, false, bf1.v,
                                                        (short)0, acc11, false, false);
    }
    // C/D layout: VGPR r, lanes 0-15 -> M=r, lanes 16-31 -> M=8+r, N = lane&15
    const float bv0 = bias[n0 + lm];
    const float bv1 = bias[n1 + lm];
    #pragma unroll
    for (int r = 0; r < 8; ++r) {
        const size_t row0 = (size_t)(mbase + hf * 8 + r);
        const size_t row1 = row0 + 16;
        C[row0 * HID + n0 + lm] = acc00[r] + bv0;
        C[row0 * HID + n1 + lm] = acc01[r] + bv1;
        C[row1 * HID + n0 + lm] = acc10[r] + bv0;
        C[row1 * HID + n1 + lm] = acc11[r] + bv1;
    }
}

// ---------------------------------------------------------------------------
// Edge kernels (wave32 per edge; lane l owns channels [8l, 8l+8), head = l>>3)
// ---------------------------------------------------------------------------
__device__ __forceinline__ void atomicMaxFloat(float* addr, float v) {
    if (v >= 0.0f) atomicMax((int*)addr, __float_as_int(v));
    else           atomicMin((unsigned int*)addr, __float_as_uint(v));
}

__global__ void edge_logits_kernel(const float* __restrict__ fs, const float* __restrict__ fd,
                                   const int* __restrict__ src, const int* __restrict__ dst,
                                   const float* __restrict__ attn,
                                   float* __restrict__ logits, float* __restrict__ mbuf) {
    const int e    = (blockIdx.x * blockDim.x + threadIdx.x) >> 5;   // EE waves total
    const int lane = threadIdx.x & 31;
    if (e >= EE) return;
    const int s = src[e], d = dst[e];
    const int c = lane * 8;
    const int head = lane >> 3;

    F8 xs, xd, aw;
    const float4* pfs = (const float4*)(fs + (size_t)s * HID + c);
    const float4* pfd = (const float4*)(fd + (size_t)d * HID + c);
    const float4* pa  = (const float4*)(attn + head * DHEAD + (lane & 7) * 8);
    xs.v[0] = pfs[0]; xs.v[1] = pfs[1];
    xd.v[0] = pfd[0]; xd.v[1] = pfd[1];
    aw.v[0] = pa[0];  aw.v[1] = pa[1];

    float v = 0.0f;
    #pragma unroll
    for (int j = 0; j < 8; ++j) {
        float t = xs.f[j] + xd.f[j];
        t = (t > 0.0f) ? t : 0.2f * t;          // leaky_relu(0.2)
        v = fmaf(t, aw.f[j], v);
    }
    // reduce over the 8 lanes of this head
    v += __shfl_xor(v, 1, 32);
    v += __shfl_xor(v, 2, 32);
    v += __shfl_xor(v, 4, 32);
    if ((lane & 7) == 0) {
        logits[(size_t)e * NHEAD + head] = v;
        atomicMaxFloat(&mbuf[(size_t)d * NHEAD + head], v);
    }
}

// exp + denom accumulation; rewrites logits[i] with exp(logit - m[dst]) so the
// aggregate pass only needs a divide (one writer per element -> race free)
__global__ void edge_denom_kernel(float* __restrict__ logits, const int* __restrict__ dst,
                                  const float* __restrict__ mbuf, float* __restrict__ denom) {
    const int i = blockIdx.x * blockDim.x + threadIdx.x;             // EE*4 threads
    if (i >= EE * NHEAD) return;
    const int e = i >> 2, head = i & 3;
    const int d = dst[e];
    const float ex = __expf(logits[i] - mbuf[(size_t)d * NHEAD + head]);
    logits[i] = ex;
    unsafeAtomicAdd(&denom[(size_t)d * NHEAD + head], ex);
}

__global__ void edge_aggregate_kernel(const float* __restrict__ fs, const float* __restrict__ ex,
                                      const int* __restrict__ src, const int* __restrict__ dst,
                                      const float* __restrict__ denom, float* __restrict__ outb) {
    const int e    = (blockIdx.x * blockDim.x + threadIdx.x) >> 5;   // EE waves total
    const int lane = threadIdx.x & 31;
    if (e >= EE) return;
    const int s = src[e], d = dst[e];
    const int c = lane * 8;
    const int head = lane >> 3;
    const float alpha = ex[(size_t)e * NHEAD + head] / denom[(size_t)d * NHEAD + head];

    const float4* p = (const float4*)(fs + (size_t)s * HID + c);
    float* o = outb + (size_t)d * HID + c;
    F8 x; x.v[0] = p[0]; x.v[1] = p[1];
    #pragma unroll
    for (int j = 0; j < 8; ++j) unsafeAtomicAdd(o + j, alpha * x.f[j]);
}

// ---------------------------------------------------------------------------
// Host side
// ---------------------------------------------------------------------------
extern "C" void kernel_launch(void* const* d_in, const int* in_sizes, int n_in,
                              void* d_out, int out_size, void* d_ws, size_t ws_size,
                              hipStream_t stream) {
    const float* atom  = (const float*)d_in[0];
    const int*   src   = (const int*)  d_in[1];
    const int*   dst   = (const int*)  d_in[2];
    const float* W_in  = (const float*)d_in[3];
    const float* b_in  = (const float*)d_in[4];
    const float* W_src = (const float*)d_in[5];
    const float* b_src = (const float*)d_in[6];
    const float* W_dst = (const float*)d_in[7];
    const float* b_dst = (const float*)d_in[8];
    const float* attn  = (const float*)d_in[9];
    const float* bias  = (const float*)d_in[10];
    float* out = (float*)d_out;

    // carve workspace (256B-aligned regions)
    char* ws = (char*)d_ws;
    size_t off = 0;
    auto carve = [&](size_t bytes) -> void* {
        void* p = ws + off;
        off += (bytes + 255) & ~(size_t)255;
        return p;
    };
    __bf16* abf    = (__bf16*)carve((size_t)NN * KIN * 2);      // padded atom, bf16
    __bf16* hbf    = (__bf16*)carve((size_t)NN * HID * 2);      // h in bf16
    float*  fs     = (float*) carve((size_t)NN * HID * 4);      // src projection
    float*  fd     = (float*) carve((size_t)NN * HID * 4);      // dst projection / h fp32 scratch
    __bf16* WtS    = (__bf16*)carve((size_t)HID * HID * 2);     // transposed bf16 weights
    __bf16* WtD    = (__bf16*)carve((size_t)HID * HID * 2);
    float*  logits = (float*) carve((size_t)EE * NHEAD * 4);
    float*  mbuf   = (float*) carve((size_t)NN * NHEAD * 4);
    float*  denom  = (float*) carve((size_t)NN * NHEAD * 4);

    const int B = 256;

    // input projection: h0 = atom @ W_in + b_in   (into fd as fp32 scratch)
    pad_atom_kernel<<<NN * KIN / B, B, 0, stream>>>(atom, abf);
    prep_wt_kernel<<<HID * KIN / B, B, 0, stream>>>(W_in, WtS, AD, KIN);
    gemm_bias_wmma<<<NN / 32, B, 0, stream>>>(abf, WtS, b_in, fd, KIN);
    cast_bf16_kernel<<<NN * HID / B, B, 0, stream>>>(fd, hbf);

    for (int l = 0; l < 3; ++l) {
        prep_wt_kernel<<<HID * HID / B, B, 0, stream>>>(W_src + (size_t)l * HID * HID, WtS, HID, HID);
        prep_wt_kernel<<<HID * HID / B, B, 0, stream>>>(W_dst + (size_t)l * HID * HID, WtD, HID, HID);
        gemm_bias_wmma<<<NN / 32, B, 0, stream>>>(hbf, WtS, b_src + (size_t)l * HID, fs, HID);
        gemm_bias_wmma<<<NN / 32, B, 0, stream>>>(hbf, WtD, b_dst + (size_t)l * HID, fd, HID);

        init_layer_kernel<<<NN * HID / B, B, 0, stream>>>(mbuf, denom, out, bias + (size_t)l * HID);
        edge_logits_kernel<<<EE / 8, B, 0, stream>>>(fs, fd, src, dst,
                                                     attn + (size_t)l * NHEAD * DHEAD, logits, mbuf);
        edge_denom_kernel<<<EE * NHEAD / B, B, 0, stream>>>(logits, dst, mbuf, denom);
        edge_aggregate_kernel<<<EE / 8, B, 0, stream>>>(fs, logits, src, dst, denom, out);

        if (l < 2) cast_bf16_kernel<<<NN * HID / B, B, 0, stream>>>(out, hbf);
    }
    (void)in_sizes; (void)n_in; (void)out_size; (void)ws_size;
}
